// SelfAttention_549755814600
// MI455X (gfx1250) — matrix operand order repaired and checked by reading, not compiled, not run on hardware
//
#include <hip/hip_runtime.h>
#include <stdint.h>

// ---------------------------------------------------------------------------
// Fused multi-head self-attention for MI455X (gfx1250, wave32, WMMA).
// B=4, S=2048, E=1024, H=16, D=64.  bf16 WMMA (16x16x32), fp32 accumulate.
// Pipeline: proj(Q,K) -> proj(V, transposed) -> flash-attn (async LDS ping-
// pong) -> output GEMM (all-bf16 operands).
// ---------------------------------------------------------------------------

#define B_  4
#define S_  2048
#define E_  1024
#define H_  16
#define D_  64
#define KT  64          // keys per flash tile
#define LDK 72          // LDS row pitch (bf16 elems); 144B = 9*16B, keeps 16B align

typedef __attribute__((ext_vector_type(16))) __bf16 v16bf;
typedef __attribute__((ext_vector_type(8)))  float  v8f;
typedef __attribute__((ext_vector_type(4)))  int    v4i;

#if defined(__AMDGCN__) && \
    __has_builtin(__builtin_amdgcn_global_load_async_to_lds_b128) && \
    __has_builtin(__builtin_amdgcn_s_wait_asynccnt)
#define HAS_ASYNC_LDS 1
typedef __attribute__((address_space(1))) v4i as1_v4i;  // global 16B vector
typedef __attribute__((address_space(3))) v4i as3_v4i;  // LDS 16B vector
#else
#define HAS_ASYNC_LDS 0
#endif

// --- CDNA5 wave32 WMMA fragment index maps (CDNA5 ISA §7.12.2) ---
// A (16x32, bf16): lane L holds row M=L%16; element j -> K = 16*half + j (this code's order),
//   half = L>=16.
// B (32x16, bf16): lane L holds col N=L%16; element j -> K = 16*half + j (this code's order, as A).
// C/D (16x16, f32): lane L element r -> row M = r + 8*half, col N = L%16.
static __device__ __forceinline__ int a_kidx(int j, int half) {
  return half * 16 + j;
}

static __device__ __forceinline__ v16bf load_a_f32(const float* t, int ld, int k0,
                                                   int l16, int half) {
  v16bf a;
#pragma unroll
  for (int j = 0; j < 16; ++j) a[j] = (__bf16)t[l16 * ld + k0 + a_kidx(j, half)];
  return a;
}
static __device__ __forceinline__ v16bf load_a_bf16(const __bf16* t, int ld, int k0,
                                                    int l16, int half) {
  v16bf a;
#pragma unroll
  for (int j = 0; j < 16; ++j) a[j] = t[l16 * ld + k0 + a_kidx(j, half)];
  return a;
}
static __device__ __forceinline__ v16bf load_b_f32(const float* t, int ld, int k0,
                                                   int l16, int half) {
  v16bf b;
#pragma unroll
  for (int j = 0; j < 16; ++j) b[j] = (__bf16)t[l16 * ld + k0 + half * 16 + j];
  return b;
}
static __device__ __forceinline__ v16bf load_b_bf16(const __bf16* t, int ld, int k0,
                                                    int l16, int half) {
  v16bf b;
#pragma unroll
  for (int j = 0; j < 16; ++j) b[j] = t[l16 * ld + k0 + half * 16 + j];
  return b;
}

static __device__ __forceinline__ v8f wmma_bf16(v16bf a, v16bf b, v8f c) {
  return __builtin_amdgcn_wmma_f32_16x16x32_bf16(false, a, false, b, (short)0, c,
                                                 false, false);
}

static __device__ __forceinline__ v8f zero8() {
  v8f z = {0.f, 0.f, 0.f, 0.f, 0.f, 0.f, 0.f, 0.f};
  return z;
}

// --- async (or fallback) 16-byte global -> LDS copy -------------------------
static __device__ __forceinline__ void copy16B_to_lds(const __bf16* g, __bf16* l) {
#if HAS_ASYNC_LDS
  __builtin_amdgcn_global_load_async_to_lds_b128((as1_v4i*)g, (as3_v4i*)l, 0, 0);
#else
  *(uint4*)l = *(const uint4*)g;
#endif
}
static __device__ __forceinline__ void async_wait_all() {
#if HAS_ASYNC_LDS
  __builtin_amdgcn_s_wait_asynccnt(0);
#endif
}
// Stage a 64-row x 64-col bf16 tile (row pitch srcPitch elems) into LDS
// (pitch LDK). 256 threads, each thread moves 2 x 16B.
static __device__ __forceinline__ void stage64(const __bf16* __restrict__ src,
                                               int srcPitch, __bf16* dst, int t) {
  const int row = t >> 3;        // 0..31
  const int off = (t & 7) * 8;   // elem offset: 0,8,...,56 (16B granules)
#pragma unroll
  for (int rr = 0; rr < 2; ++rr) {
    const int r = row + rr * 32;
    copy16B_to_lds(src + (size_t)r * srcPitch + off, dst + r * LDK + off);
  }
}

// ---------------------------------------------------------------------------
// Kernel 1: per-head projection  Y = X @ W^T (bf16 out).
// transposed=0: Y[b,h,s,d]  (pitch D);  transposed=1: Y[b,h,d,s] (pitch S).
// One wave = one 16-row x 64-col tile. Scale folded in (Q: 2^-5, exact bf16).
// ---------------------------------------------------------------------------
__global__ __launch_bounds__(256) void proj_kernel(const float* __restrict__ X,
                                                   const float* __restrict__ W,
                                                   __bf16* __restrict__ Y,
                                                   float scale, int transposed) {
  const int lane = threadIdx.x & 31;
  const int w    = threadIdx.x >> 5;
  const int half = lane >> 4;
  const int l16  = lane & 15;

  const int tile = blockIdx.x * 8 + w;       // 0 .. B*H*(S/16)-1 = 8191
  const int s16  = tile & (S_ / 16 - 1);     // 0..127
  const int bh   = tile >> 7;
  const int b    = bh >> 4;
  const int h    = bh & 15;

  const float* Xb = X + ((size_t)(b * S_ + s16 * 16)) * E_ + h * D_;

  v16bf a[2];
#pragma unroll
  for (int kc = 0; kc < 2; ++kc) a[kc] = load_a_f32(Xb, E_, kc * 32, l16, half);
  if (scale != 1.0f) {
#pragma unroll
    for (int kc = 0; kc < 2; ++kc)
#pragma unroll
      for (int j = 0; j < 16; ++j) a[kc][j] = (__bf16)((float)a[kc][j] * scale);
  }

#pragma unroll
  for (int nd = 0; nd < 4; ++nd) {
    v8f acc = zero8();
#pragma unroll
    for (int kc = 0; kc < 2; ++kc) {
      v16bf bb = load_b_f32(W + (size_t)(nd * 16) * D_, D_, kc * 32, l16, half);
      acc = wmma_bf16(a[kc], bb, acc);
    }
    if (!transposed) {
      __bf16* Yb = Y + ((size_t)(b * H_ + h) * S_ + s16 * 16) * D_;
#pragma unroll
      for (int r = 0; r < 8; ++r)
        Yb[(size_t)(r + half * 8) * D_ + nd * 16 + l16] = (__bf16)acc[r];
    } else {
      // Y[b,h,d,s]: for fixed (nd,l16), r indexes 8 consecutive s -> contiguous
      __bf16* Yb = Y + ((size_t)(b * H_ + h) * D_ + nd * 16 + l16) * S_ +
                   s16 * 16 + half * 8;
#pragma unroll
      for (int r = 0; r < 8; ++r) Yb[r] = (__bf16)acc[r];
    }
  }
}

// ---------------------------------------------------------------------------
// Kernel 2: flash attention. Block = (b, h, 128-query slab); 8 waves x 16 rows.
// K tiles ([key][d]) and V^T tiles ([d][key]) async-staged into double-
// buffered LDS; online softmax with wave32 shfl reductions; P round-trips
// through wave-private LDS (C-layout -> A-layout). Output bf16 [B,S,E].
// ---------------------------------------------------------------------------
__global__ __launch_bounds__(256) void attn_kernel(const __bf16* __restrict__ Qp,
                                                   const __bf16* __restrict__ Kp,
                                                   const __bf16* __restrict__ VpT,
                                                   __bf16* __restrict__ Xbf) {
  __shared__ __bf16 Ktile[2][KT * LDK];   // [key][d]
  __shared__ __bf16 Vtile[2][D_ * LDK];   // [d][key]
  __shared__ __bf16 Pt[8][16 * LDK];      // per-wave P, [qrow][key]

  const int t    = threadIdx.x;
  const int lane = t & 31;
  const int w    = t >> 5;
  const int half = lane >> 4;
  const int l16  = lane & 15;

  const int qb = blockIdx.x & 15;         // 16 query slabs of 128
  const int bh = blockIdx.x >> 4;
  const int b  = bh >> 4;
  const int h  = bh & 15;

  const __bf16* Qb = Qp  + (size_t)(b * H_ + h) * S_ * D_;  // [s][d] pitch 64
  const __bf16* Kb = Kp  + (size_t)(b * H_ + h) * S_ * D_;  // [s][d] pitch 64
  const __bf16* Vb = VpT + (size_t)(b * H_ + h) * D_ * S_;  // [d][s] pitch 2048
  const int q0 = qb * 128 + w * 16;

  // Q fragments (already pre-scaled by 1/sqrt(E) in proj)
  v16bf qa[2];
#pragma unroll
  for (int kc = 0; kc < 2; ++kc)
    qa[kc] = load_a_bf16(Qb + (size_t)q0 * D_, D_, kc * 32, l16, half);

  v8f o[4];
#pragma unroll
  for (int nd = 0; nd < 4; ++nd) o[nd] = zero8();
  float m_[8], l_[8];
#pragma unroll
  for (int r = 0; r < 8; ++r) { m_[r] = -1e30f; l_[r] = 0.f; }

  // prologue: stage tile 0 into buffer 0
  stage64(Kb, D_, Ktile[0], t);
  stage64(Vb, S_, Vtile[0], t);
  async_wait_all();
  __syncthreads();

  int buf = 0;
  for (int k0 = 0; k0 < S_; k0 += KT) {
    // prefetch next tile into the other buffer while computing this one
    if (k0 + KT < S_) {
      stage64(Kb + (size_t)(k0 + KT) * D_, D_, Ktile[buf ^ 1], t);
      stage64(Vb + (k0 + KT),            S_, Vtile[buf ^ 1], t);
    }

    // S = Q K^T : 4 N-tiles (keys) x 2 K-chunks (d)
    v8f s[4];
#pragma unroll
    for (int n = 0; n < 4; ++n) {
      s[n] = zero8();
#pragma unroll
      for (int kc = 0; kc < 2; ++kc) {
        v16bf bb = load_b_bf16(&Ktile[buf][n * 16 * LDK], LDK, kc * 32, l16, half);
        s[n] = wmma_bf16(qa[kc], bb, s[n]);
      }
    }

    // online softmax per row (C rows live in 16-lane halves)
#pragma unroll
    for (int r = 0; r < 8; ++r) {
      float mx = fmaxf(fmaxf(s[0][r], s[1][r]), fmaxf(s[2][r], s[3][r]));
      mx = fmaxf(mx, __shfl_xor(mx, 1, 32));
      mx = fmaxf(mx, __shfl_xor(mx, 2, 32));
      mx = fmaxf(mx, __shfl_xor(mx, 4, 32));
      mx = fmaxf(mx, __shfl_xor(mx, 8, 32));
      const float mn = fmaxf(m_[r], mx);
      const float alpha = __expf(m_[r] - mn);
      m_[r] = mn;
      float rs = 0.f;
#pragma unroll
      for (int n = 0; n < 4; ++n) {
        float e = __expf(s[n][r] - mn);
        s[n][r] = e;
        rs += e;
      }
      rs += __shfl_xor(rs, 1, 32);
      rs += __shfl_xor(rs, 2, 32);
      rs += __shfl_xor(rs, 4, 32);
      rs += __shfl_xor(rs, 8, 32);
      l_[r] = l_[r] * alpha + rs;
#pragma unroll
      for (int n = 0; n < 4; ++n) o[n][r] *= alpha;
    }

    // P: C-layout -> wave-private LDS -> A-layout fragments
#pragma unroll
    for (int n = 0; n < 4; ++n)
#pragma unroll
      for (int r = 0; r < 8; ++r)
        Pt[w][(r + half * 8) * LDK + n * 16 + l16] = (__bf16)s[n][r];

    // O += P V : K-dim = keys (2 chunks of 32), N-dim = d (4 tiles of 16)
#pragma unroll
    for (int kc = 0; kc < 2; ++kc) {
      v16bf pa = load_a_bf16(&Pt[w][0], LDK, kc * 32, l16, half);
#pragma unroll
      for (int nd = 0; nd < 4; ++nd) {
        v16bf vb = load_b_bf16(&Vtile[buf][nd * 16 * LDK], LDK, kc * 32, l16, half);
        o[nd] = wmma_bf16(pa, vb, o[nd]);
      }
    }

    async_wait_all();     // next tile fully in LDS (this wave's issues)
    __syncthreads();      // ... and everyone else's; prev readers done
    buf ^= 1;
  }

  // epilogue: O /= l, write bf16 [B, S, E] (feeds bf16 GEMM directly)
  float inv[8];
#pragma unroll
  for (int r = 0; r < 8; ++r) inv[r] = 1.0f / l_[r];
#pragma unroll
  for (int nd = 0; nd < 4; ++nd)
#pragma unroll
    for (int r = 0; r < 8; ++r) {
      const int srow = q0 + r + half * 8;
      Xbf[((size_t)(b * S_ + srow)) * E_ + h * D_ + nd * 16 + l16] =
          (__bf16)(o[nd][r] * inv[r]);
    }
}

// ---------------------------------------------------------------------------
// Kernel 3a: one-shot fp32 -> bf16 conversion (for Wo)
// ---------------------------------------------------------------------------
__global__ __launch_bounds__(256) void cvt_bf16_kernel(const float* __restrict__ in,
                                                       __bf16* __restrict__ outp,
                                                       int n) {
  const int i = (blockIdx.x * 256 + threadIdx.x) * 8;
  if (i + 8 <= n) {
#pragma unroll
    for (int j = 0; j < 8; ++j) outp[i + j] = (__bf16)in[i + j];
  }
}

// ---------------------------------------------------------------------------
// Kernel 3b: output projection  Y = X @ Wo^T + bo  (8192 x 1024 x 1024)
// All-bf16 operands (pre-converted), fp32 accumulate + fp32 store.
// ---------------------------------------------------------------------------
__global__ __launch_bounds__(256) void outproj_kernel(const __bf16* __restrict__ X,
                                                      const __bf16* __restrict__ Wob,
                                                      const float* __restrict__ bo,
                                                      float* __restrict__ Y) {
  const int lane = threadIdx.x & 31;
  const int w    = threadIdx.x >> 5;
  const int half = lane >> 4;
  const int l16  = lane & 15;

  const int tile = blockIdx.x * 8 + w;   // 0..8191
  const int ct   = tile & 15;            // 16 col tiles of 64
  const int rt   = tile >> 4;            // 512 row tiles of 16

  const __bf16* Xb = X + (size_t)(rt * 16) * E_;

  v8f acc[4];
#pragma unroll
  for (int nd = 0; nd < 4; ++nd) acc[nd] = zero8();

  for (int kc = 0; kc < E_; kc += 32) {
    v16bf a = load_a_bf16(Xb, E_, kc, l16, half);
#pragma unroll
    for (int nd = 0; nd < 4; ++nd) {
      v16bf bb = load_b_bf16(Wob + (size_t)(ct * 64 + nd * 16) * E_, E_, kc,
                             l16, half);
      acc[nd] = wmma_bf16(a, bb, acc[nd]);
    }
  }

#pragma unroll
  for (int nd = 0; nd < 4; ++nd)
#pragma unroll
    for (int r = 0; r < 8; ++r) {
      const int col = ct * 64 + nd * 16 + l16;
      Y[(size_t)(rt * 16 + r + half * 8) * E_ + col] = acc[nd][r] + bo[col];
    }
}

// ---------------------------------------------------------------------------
extern "C" void kernel_launch(void* const* d_in, const int* in_sizes, int n_in,
                              void* d_out, int out_size, void* d_ws, size_t ws_size,
                              hipStream_t stream) {
  (void)in_sizes; (void)n_in; (void)out_size; (void)ws_size;
  const float* values = (const float*)d_in[0];
  const float* keys   = (const float*)d_in[1];
  const float* query  = (const float*)d_in[2];
  const float* Wv     = (const float*)d_in[3];
  const float* Wk     = (const float*)d_in[4];
  const float* Wq     = (const float*)d_in[5];
  const float* Wo     = (const float*)d_in[6];
  const float* bo     = (const float*)d_in[7];
  float* out = (float*)d_out;

  // ws layout (all bf16): Qp[B,H,S,D] 16MB | Kp 16MB | VpT[B,H,D,S] 16MB |
  // Xbf[B*S,E] 16MB | Wobf[E,E] 2MB   (total 66MB)
  const size_t NP = (size_t)B_ * H_ * S_ * D_;     // 8,388,608 elems
  __bf16* Qp   = (__bf16*)d_ws;
  __bf16* Kp   = Qp + NP;
  __bf16* VpT  = Kp + NP;
  __bf16* Xbf  = VpT + NP;
  __bf16* Wobf = Xbf + (size_t)B_ * S_ * E_;

  const float qscale = 0.03125f;  // 1/sqrt(EMBED) = 2^-5, exact in bf16

  dim3 blk(256);
  proj_kernel<<<dim3(1024), blk, 0, stream>>>(query,  Wq, Qp,  qscale, 0);
  proj_kernel<<<dim3(1024), blk, 0, stream>>>(keys,   Wk, Kp,  1.0f,   0);
  proj_kernel<<<dim3(1024), blk, 0, stream>>>(values, Wv, VpT, 1.0f,   1);
  cvt_bf16_kernel<<<dim3((E_ * E_) / (256 * 8)), blk, 0, stream>>>(Wo, Wobf,
                                                                   E_ * E_);
  attn_kernel<<<dim3(1024), blk, 0, stream>>>(Qp, Kp, VpT, Xbf);
  outproj_kernel<<<dim3(1024), blk, 0, stream>>>(Xbf, Wobf, bo, out);
}
